// WaveNet_55027120996882
// MI455X (gfx1250) — matrix operand order repaired and checked by reading, not compiled, not run on hardware
//
#include <hip/hip_runtime.h>
#include <hip/hip_bf16.h>

typedef __attribute__((ext_vector_type(16))) _Float16 v16h;
typedef __attribute__((ext_vector_type(8)))  _Float16 v8h;
typedef __attribute__((ext_vector_type(8)))  float    v8f;

#define BATCH 4
#define TLEN  32000
#define NLAYERS 30

// ---------------------------------------------------------------------------
// Fast device activations: hardware transcendental where available.
// ---------------------------------------------------------------------------
__device__ __forceinline__ float fast_tanh(float x) {
#if __has_builtin(__builtin_amdgcn_tanhf)
    return __builtin_amdgcn_tanhf(x);
#elif __has_builtin(__builtin_amdgcn_tanh_f32)
    return __builtin_amdgcn_tanh_f32(x);
#else
    float e = __expf(-2.0f * x);
    return (1.0f - e) / (1.0f + e);
#endif
}
__device__ __forceinline__ float fast_sigmoid(float x) {
    return 1.0f / (1.0f + __expf(-x));
}

// ---------------------------------------------------------------------------
// Weight pre-swizzle into WMMA A-fragment order.
// A-fragment (16x32 f16): lane l holds row M = l%16; K values
//   halfs j=0..15 -> K = kc*32 + 16*(l/16) + j
// ---------------------------------------------------------------------------
__device__ __forceinline__ int frag_k(int kc, int lane, int j) {
    return kc * 32 + (lane >> 4) * 16 + j;
}

// Gate weights: M=128, K=224 (0-63: tap0 conv_w[..][k][0], 64-127: tap1,
// 128-207: cond_w, 208-223: zero pad). Layout [L][8 mt][7 kc][32 lane][16 j].
__global__ void pack_gate_kernel(const float* __restrict__ conv_w,
                                 const float* __restrict__ cond_w,
                                 _Float16* __restrict__ dst) {
    const int PER_L = 8 * 7 * 32 * 16;
    int idx = blockIdx.x * blockDim.x + threadIdx.x;
    if (idx >= NLAYERS * PER_L) return;
    int l = idx / PER_L, r = idx % PER_L;
    int mt = r / (7 * 32 * 16); r %= 7 * 32 * 16;
    int kc = r / (32 * 16);     r %= 32 * 16;
    int lane = r / 16, j = r % 16;
    int M = mt * 16 + (lane & 15);
    int k = frag_k(kc, lane, j);
    float v = 0.0f;
    if (k < 64)        v = conv_w[((size_t)(l * 128 + M) * 64 + k) * 2 + 0];
    else if (k < 128)  v = conv_w[((size_t)(l * 128 + M) * 64 + (k - 64)) * 2 + 1];
    else if (k < 208)  v = cond_w[(size_t)(l * 128 + M) * 80 + (k - 128)];
    dst[idx] = (_Float16)v;
}

// Generic [O][kdim] row-major f32 -> [L][mtiles][kchunks][32][16] fragments.
__global__ void pack_mat_kernel(const float* __restrict__ src,
                                _Float16* __restrict__ dst,
                                int nLayers, int mtiles, int kchunks,
                                int layerStride) {
    int PER_L = mtiles * kchunks * 32 * 16;
    int idx = blockIdx.x * blockDim.x + threadIdx.x;
    if (idx >= nLayers * PER_L) return;
    int l = idx / PER_L, r = idx % PER_L;
    int mt = r / (kchunks * 32 * 16); r %= kchunks * 32 * 16;
    int kc = r / (32 * 16);           r %= 32 * 16;
    int lane = r / 16, j = r % 16;
    int M = mt * 16 + (lane & 15);
    int k = frag_k(kc, lane, j);
    int kdim = kchunks * 32;
    dst[idx] = (_Float16)src[(size_t)l * layerStride + M * kdim + k];
}

// ---------------------------------------------------------------------------
// Conditioning path (tiny): conv_in 1x1, repeat x10 + 21-tap avg,
// repeat x8 + 17-tap avg. Output cfull [B][T][96] f16 (ch 80..95 zero).
// ---------------------------------------------------------------------------
__global__ void conv_in_kernel(const float* __restrict__ c,
                               const float* __restrict__ w,
                               float* __restrict__ cc) {
    int idx = blockIdx.x * blockDim.x + threadIdx.x;
    if (idx >= BATCH * 80 * 400) return;
    int f = idx % 400, o = (idx / 400) % 80, b = idx / (400 * 80);
    float s = 0.0f;
    for (int ic = 0; ic < 80; ++ic)
        s += w[o * 80 + ic] * c[(size_t)(b * 80 + ic) * 400 + f];
    cc[idx] = s;
}

__global__ void upsample1_kernel(const float* __restrict__ cc,
                                 const float* __restrict__ w0,
                                 float* __restrict__ u1) {
    int idx = blockIdx.x * blockDim.x + threadIdx.x;
    if (idx >= BATCH * 80 * 4000) return;
    int t = idx % 4000, ch = (idx / 4000) % 80, b = idx / (4000 * 80);
    float s = 0.0f;
    for (int k = 0; k < 21; ++k) {
        int r = t - 10 + k;
        if (r >= 0 && r < 4000)
            s += w0[k] * cc[(size_t)(b * 80 + ch) * 400 + r / 10];
    }
    u1[idx] = s;
}

__global__ void upsample2_kernel(const float* __restrict__ u1,
                                 const float* __restrict__ w1,
                                 _Float16* __restrict__ cfull) {
    int idx = blockIdx.x * blockDim.x + threadIdx.x;
    if (idx >= BATCH * TLEN * 96) return;
    int ch = idx % 96, t = (idx / 96) % TLEN, b = idx / (96 * TLEN);
    float s = 0.0f;
    if (ch < 80) {
        for (int k = 0; k < 17; ++k) {
            int r = t - 8 + k;
            if (r >= 0 && r < TLEN)
                s += w1[k] * u1[(size_t)(b * 80 + ch) * 4000 + r / 8];
        }
    }
    cfull[idx] = (_Float16)s;  // idx == ((b*T + t)*96 + ch)
}

// ---------------------------------------------------------------------------
// Transpose x [B,256,T] f32 -> xt [B][T][256] f16 via LDS 32x32 tiles.
// ---------------------------------------------------------------------------
__global__ __launch_bounds__(256) void transpose_x_kernel(
    const float* __restrict__ x, _Float16* __restrict__ xt) {
    __shared__ float tile[32][33];
    int b  = blockIdx.z;
    int t0 = blockIdx.x * 32;
    int c0 = blockIdx.y * 32;
    int tx = threadIdx.x;        // 0..31
    int ty = threadIdx.y;        // 0..7
    #pragma unroll
    for (int i = 0; i < 32; i += 8)
        tile[ty + i][tx] = x[((size_t)(b * 256 + c0 + ty + i)) * TLEN + t0 + tx];
    __syncthreads();
    #pragma unroll
    for (int i = 0; i < 32; i += 8) {
        int t = t0 + ty + i;
        xt[((size_t)b * TLEN + t) * 256 + c0 + tx] = (_Float16)tile[tx][ty + i];
    }
}

// ---------------------------------------------------------------------------
// First 1x1 conv as WMMA GEMM: h0 = first_w(64x256) @ xt^T + first_b.
// Block = 128 threads (4 waves), each wave one 16-column time tile.
// ---------------------------------------------------------------------------
__global__ __launch_bounds__(128) void first_conv_wmma_kernel(
    const _Float16* __restrict__ xt, const _Float16* __restrict__ fwp,
    const float* __restrict__ bias, _Float16* __restrict__ h) {
    const int lane = threadIdx.x & 31;
    const int wid  = threadIdx.x >> 5;
    const int col  = lane & 15;
    const int hi   = lane >> 4;
    const int b    = blockIdx.y;
    const int t    = (blockIdx.x * 4 + wid) * 16 + col;
    const size_t rowX = ((size_t)b * TLEN + t) * 256;
    const size_t rowH = ((size_t)b * TLEN + t) * 64;

    v16h bf[8];
    #pragma unroll
    for (int kc = 0; kc < 8; ++kc)
        bf[kc] = *(const v16h*)(xt + rowX + kc * 32 + hi * 16);

    #pragma unroll
    for (int mt = 0; mt < 4; ++mt) {
        v8f a = {};
        #pragma unroll
        for (int kc = 0; kc < 8; ++kc) {
            v16h af = *(const v16h*)(fwp + ((size_t)(mt * 8 + kc) * 32 + lane) * 16);
            a = __builtin_amdgcn_wmma_f32_16x16x32_f16(
                    false, af, false, bf[kc], (short)0, a, false, false);
        }
        int cb = mt * 16 + hi * 8;
        v8h hv;
        #pragma unroll
        for (int v = 0; v < 8; ++v)
            hv[v] = (_Float16)(a[v] + bias[cb + v]);
        *(v8h*)(h + rowH + cb) = hv;
    }
}

// ---------------------------------------------------------------------------
// WaveNet residual layer. Block = 128 threads (4 waves), each wave owns one
// 16-column (time) tile. Gate GEMM M=128, K=224; then gate nonlinearity;
// LDS transpose C-layout -> B-layout; skip/out GEMMs M=64, K=64.
// ---------------------------------------------------------------------------
__global__ __launch_bounds__(128) void layer_kernel(
    const _Float16* __restrict__ h_in, _Float16* __restrict__ h_out,
    const _Float16* __restrict__ cfull, float* __restrict__ skip,
    const _Float16* __restrict__ wg, const _Float16* __restrict__ wsp,
    const _Float16* __restrict__ wop, const float* __restrict__ conv_b,
    const float* __restrict__ skip_b, const float* __restrict__ out_b,
    int d) {
    __shared__ _Float16 zbuf[4][16][64] __attribute__((aligned(64)));
    const int lane = threadIdx.x & 31;
    const int wid  = threadIdx.x >> 5;
    const int col  = lane & 15;
    const int hi   = lane >> 4;
    const int b    = blockIdx.y;
    const int t    = (blockIdx.x * 4 + wid) * 16 + col;
    const size_t rowH = ((size_t)b * TLEN + t) * 64;
    const size_t rowC = ((size_t)b * TLEN + t) * 96;

    // B fragments: K chunks 0-1 = h(t-d), 2-3 = h(t), 4-6 = cond(t)
    v16h bf[7];
    bf[0] = v16h{}; bf[1] = v16h{};
    int tm = t - d;
    if (tm >= 0) {
        size_t rowM = ((size_t)b * TLEN + tm) * 64;
        bf[0] = *(const v16h*)(h_in + rowM + hi * 16);
        bf[1] = *(const v16h*)(h_in + rowM + 32 + hi * 16);
    }
    bf[2] = *(const v16h*)(h_in + rowH + hi * 16);
    bf[3] = *(const v16h*)(h_in + rowH + 32 + hi * 16);
    bf[4] = *(const v16h*)(cfull + rowC + hi * 16);
    bf[5] = *(const v16h*)(cfull + rowC + 32 + hi * 16);
    bf[6] = *(const v16h*)(cfull + rowC + 64 + hi * 16);

    // Gate GEMM: 8 M-tiles x 7 K-chunks
    v8f acc[8];
    #pragma unroll
    for (int mt = 0; mt < 8; ++mt) {
        v8f a = {};
        #pragma unroll
        for (int kc = 0; kc < 7; ++kc) {
            v16h af = *(const v16h*)(wg + ((size_t)(mt * 7 + kc) * 32 + lane) * 16);
            a = __builtin_amdgcn_wmma_f32_16x16x32_f16(
                    false, af, false, bf[kc], (short)0, a, false, false);
        }
        acc[mt] = a;
    }

    // Gate nonlinearity: z = tanh(a) * sigmoid(b); write to LDS (B-layout src)
    #pragma unroll
    for (int mt = 0; mt < 4; ++mt) {
        int cb = mt * 16 + hi * 8;
        v8h zp;
        #pragma unroll
        for (int v = 0; v < 8; ++v) {
            float av = acc[mt][v]     + conv_b[cb + v];
            float bv = acc[mt + 4][v] + conv_b[64 + cb + v];
            zp[v] = (_Float16)(fast_tanh(av) * fast_sigmoid(bv));
        }
        *(v8h*)&zbuf[wid][col][cb] = zp;
    }
    __syncthreads();

    v16h zf0 = *(const v16h*)&zbuf[wid][col][hi * 16];
    v16h zf1 = *(const v16h*)&zbuf[wid][col][32 + hi * 16];

    // skip/out GEMMs + residual epilogue
    #pragma unroll
    for (int mt = 0; mt < 4; ++mt) {
        v8f as = {}, ao = {};
        v16h s0 = *(const v16h*)(wsp + ((size_t)(mt * 2 + 0) * 32 + lane) * 16);
        v16h s1 = *(const v16h*)(wsp + ((size_t)(mt * 2 + 1) * 32 + lane) * 16);
        as = __builtin_amdgcn_wmma_f32_16x16x32_f16(false, s0, false, zf0, (short)0, as, false, false);
        as = __builtin_amdgcn_wmma_f32_16x16x32_f16(false, s1, false, zf1, (short)0, as, false, false);
        v16h o0 = *(const v16h*)(wop + ((size_t)(mt * 2 + 0) * 32 + lane) * 16);
        v16h o1 = *(const v16h*)(wop + ((size_t)(mt * 2 + 1) * 32 + lane) * 16);
        ao = __builtin_amdgcn_wmma_f32_16x16x32_f16(false, o0, false, zf0, (short)0, ao, false, false);
        ao = __builtin_amdgcn_wmma_f32_16x16x32_f16(false, o1, false, zf1, (short)0, ao, false, false);

        int cb = mt * 16 + hi * 8;
        size_t idx = rowH + cb;
        float* sp = skip + idx;
        v8h hv = *(const v8h*)(h_in + idx);
        v8h ho;
        #pragma unroll
        for (int v = 0; v < 8; ++v) {
            sp[v] = sp[v] + as[v] + skip_b[cb + v];
            ho[v] = (_Float16)((float)hv[v] + ao[v] + out_b[cb + v]);
        }
        *(v8h*)(h_out + idx) = ho;
    }
}

// ---------------------------------------------------------------------------
// Last stage: relu(skip) -> last1(64x64) -> relu -> last2(256x64) -> out.
// ---------------------------------------------------------------------------
__global__ __launch_bounds__(128) void last_kernel(
    const float* __restrict__ skip,
    const _Float16* __restrict__ l1p, const _Float16* __restrict__ l2p,
    const float* __restrict__ l1b, const float* __restrict__ l2b,
    float* __restrict__ out) {
    __shared__ _Float16 zbuf[4][16][64] __attribute__((aligned(64)));
    const int lane = threadIdx.x & 31;
    const int wid  = threadIdx.x >> 5;
    const int col  = lane & 15;
    const int hi   = lane >> 4;
    const int b    = blockIdx.y;
    const int t    = (blockIdx.x * 4 + wid) * 16 + col;
    const size_t rowH = ((size_t)b * TLEN + t) * 64;

    v16h bf[2];
    #pragma unroll
    for (int kc = 0; kc < 2; ++kc)
        #pragma unroll
        for (int j = 0; j < 16; ++j)
            bf[kc][j] = (_Float16)fmaxf(skip[rowH + kc * 32 + hi * 16 + j], 0.0f);

    #pragma unroll
    for (int mt = 0; mt < 4; ++mt) {
        v8f a = {};
        v16h a0 = *(const v16h*)(l1p + ((size_t)(mt * 2 + 0) * 32 + lane) * 16);
        v16h a1 = *(const v16h*)(l1p + ((size_t)(mt * 2 + 1) * 32 + lane) * 16);
        a = __builtin_amdgcn_wmma_f32_16x16x32_f16(false, a0, false, bf[0], (short)0, a, false, false);
        a = __builtin_amdgcn_wmma_f32_16x16x32_f16(false, a1, false, bf[1], (short)0, a, false, false);
        int cb = mt * 16 + hi * 8;
        v8h zp;
        #pragma unroll
        for (int v = 0; v < 8; ++v)
            zp[v] = (_Float16)fmaxf(a[v] + l1b[cb + v], 0.0f);
        *(v8h*)&zbuf[wid][col][cb] = zp;
    }
    __syncthreads();

    v16h zf0 = *(const v16h*)&zbuf[wid][col][hi * 16];
    v16h zf1 = *(const v16h*)&zbuf[wid][col][32 + hi * 16];

    for (int mt = 0; mt < 16; ++mt) {
        v8f a = {};
        v16h a0 = *(const v16h*)(l2p + ((size_t)(mt * 2 + 0) * 32 + lane) * 16);
        v16h a1 = *(const v16h*)(l2p + ((size_t)(mt * 2 + 1) * 32 + lane) * 16);
        a = __builtin_amdgcn_wmma_f32_16x16x32_f16(false, a0, false, zf0, (short)0, a, false, false);
        a = __builtin_amdgcn_wmma_f32_16x16x32_f16(false, a1, false, zf1, (short)0, a, false, false);
        #pragma unroll
        for (int v = 0; v < 8; ++v) {
            int ch = mt * 16 + hi * 8 + v;
            out[((size_t)(b * 256 + ch)) * TLEN + t] = a[v] + l2b[ch];
        }
    }
}

// ---------------------------------------------------------------------------
extern "C" void kernel_launch(void* const* d_in, const int* in_sizes, int n_in,
                              void* d_out, int out_size, void* d_ws, size_t ws_size,
                              hipStream_t stream) {
    const float* x        = (const float*)d_in[0];
    const float* c        = (const float*)d_in[1];
    const float* first_w  = (const float*)d_in[2];
    const float* first_b  = (const float*)d_in[3];
    const float* conv_w   = (const float*)d_in[4];
    const float* conv_b   = (const float*)d_in[5];
    const float* cond_w   = (const float*)d_in[6];
    const float* out_w    = (const float*)d_in[7];
    const float* out_b    = (const float*)d_in[8];
    const float* skip_w   = (const float*)d_in[9];
    const float* skip_b   = (const float*)d_in[10];
    const float* last1_w  = (const float*)d_in[11];
    const float* last1_b  = (const float*)d_in[12];
    const float* last2_w  = (const float*)d_in[13];
    const float* last2_b  = (const float*)d_in[14];
    const float* conv_inw = (const float*)d_in[15];
    const float* up_w0    = (const float*)d_in[16];
    const float* up_w1    = (const float*)d_in[17];
    float* out = (float*)d_out;

    // Workspace partition (256B aligned)
    char* ws = (char*)d_ws;
    size_t off = 0;
    auto carve = [&](size_t bytes) { char* p = ws + off; off += (bytes + 255) & ~(size_t)255; return p; };
    _Float16* wgp = (_Float16*)carve((size_t)NLAYERS * 8 * 7 * 32 * 16 * 2);
    _Float16* wsp = (_Float16*)carve((size_t)NLAYERS * 4 * 2 * 32 * 16 * 2);
    _Float16* wop = (_Float16*)carve((size_t)NLAYERS * 4 * 2 * 32 * 16 * 2);
    _Float16* l1p = (_Float16*)carve((size_t)4 * 2 * 32 * 16 * 2);
    _Float16* l2p = (_Float16*)carve((size_t)16 * 2 * 32 * 16 * 2);
    _Float16* fwp = (_Float16*)carve((size_t)4 * 8 * 32 * 16 * 2);
    _Float16* h0  = (_Float16*)carve((size_t)BATCH * TLEN * 64 * 2);
    _Float16* h1  = (_Float16*)carve((size_t)BATCH * TLEN * 64 * 2);
    _Float16* cf  = (_Float16*)carve((size_t)BATCH * TLEN * 96 * 2);
    _Float16* xt  = (_Float16*)carve((size_t)BATCH * TLEN * 256 * 2);
    float*    sk  = (float*)   carve((size_t)BATCH * TLEN * 64 * 4);
    float*    cc  = (float*)   carve((size_t)BATCH * 80 * 400 * 4);
    float*    u1  = (float*)   carve((size_t)BATCH * 80 * 4000 * 4);
    (void)ws_size; (void)in_sizes; (void)n_in; (void)out_size;

    // 1) Weight pre-swizzle into WMMA fragment order
    {
        int n = NLAYERS * 8 * 7 * 32 * 16;
        pack_gate_kernel<<<(n + 255) / 256, 256, 0, stream>>>(conv_w, cond_w, wgp);
        int m = NLAYERS * 4 * 2 * 32 * 16;
        pack_mat_kernel<<<(m + 255) / 256, 256, 0, stream>>>(skip_w, wsp, NLAYERS, 4, 2, 64 * 64);
        pack_mat_kernel<<<(m + 255) / 256, 256, 0, stream>>>(out_w, wop, NLAYERS, 4, 2, 64 * 64);
        int k1 = 4 * 2 * 32 * 16;
        pack_mat_kernel<<<(k1 + 255) / 256, 256, 0, stream>>>(last1_w, l1p, 1, 4, 2, 64 * 64);
        int k2 = 16 * 2 * 32 * 16;
        pack_mat_kernel<<<(k2 + 255) / 256, 256, 0, stream>>>(last2_w, l2p, 1, 16, 2, 256 * 64);
        int kf = 4 * 8 * 32 * 16;
        pack_mat_kernel<<<(kf + 255) / 256, 256, 0, stream>>>(first_w, fwp, 1, 4, 8, 64 * 256);
    }

    // 2) Conditioning upsample
    {
        int n0 = BATCH * 80 * 400;
        conv_in_kernel<<<(n0 + 255) / 256, 256, 0, stream>>>(c, conv_inw, cc);
        int n1 = BATCH * 80 * 4000;
        upsample1_kernel<<<(n1 + 255) / 256, 256, 0, stream>>>(cc, up_w0, u1);
        int n2 = BATCH * TLEN * 96;
        upsample2_kernel<<<(n2 + 255) / 256, 256, 0, stream>>>(u1, up_w1, cf);
    }

    // 3) Zero skip accumulator; transpose x; first conv via WMMA
    hipMemsetAsync(sk, 0, (size_t)BATCH * TLEN * 64 * 4, stream);
    {
        dim3 tgrid(TLEN / 32, 256 / 32, BATCH);
        transpose_x_kernel<<<tgrid, dim3(32, 8), 0, stream>>>(x, xt);
        dim3 fgrid(TLEN / 64, BATCH);
        first_conv_wmma_kernel<<<fgrid, 128, 0, stream>>>(xt, fwp, first_b, h0);
    }

    // 4) 30 residual layers (ping-pong h buffers)
    dim3 grid(TLEN / 64, BATCH);
    for (int l = 0; l < NLAYERS; ++l) {
        int d = 1 << (l % 10);
        const _Float16* hin = (l & 1) ? h1 : h0;
        _Float16* hout      = (l & 1) ? h0 : h1;
        layer_kernel<<<grid, 128, 0, stream>>>(
            hin, hout, cf, sk,
            wgp + (size_t)l * 8 * 7 * 32 * 16,
            wsp + (size_t)l * 4 * 2 * 32 * 16,
            wop + (size_t)l * 4 * 2 * 32 * 16,
            conv_b + l * 128, skip_b + l * 64, out_b + l * 64, d);
    }

    // 5) Output stage
    last_kernel<<<grid, 128, 0, stream>>>(sk, l1p, l2p, last1_b, last2_b, out);
}
